// PointNet_80333068304471
// MI455X (gfx1250) — matrix operand order repaired and checked by reading, not compiled, not run on hardware
//
#include <hip/hip_runtime.h>

// ---------------------------------------------------------------------------
// PointNet-style pipeline for MI455X (gfx1250, wave32, WMMA).
// Compute-bound (74 GFLOP vs ~32MB traffic) -> v_wmma_f32_16x16x32_f16 path.
// Point kernel: 2-way M register tiling (sized to stay under 256 VGPRs at
// 4 waves/SIMD); layers 1-2 run weights-in-A / activations-in-B so each WMMA
// epilogue is a packed ds_store_b128 (layer 1) or feeds layer-3's A fragments
// DIRECTLY in registers (layer 2): the packed 8-channel outputs of layer-2
// mt=2ks and mt=2ks+1 are exactly halves 0..7 / 8..15 of layer-3's A fragment
// for k-step ks. Layer 3 runs act-in-A so the segment-max reduction stays
// lane-parallel (1 shfl + 1 atomic per channel).
// ---------------------------------------------------------------------------

#define NPTS  1000000
#define NSEG  4096
#define KOUT  75          // 25*3 head outputs

typedef __attribute__((ext_vector_type(16))) _Float16 v16h;
typedef __attribute__((ext_vector_type(8)))  float    v8f;

// fragment table offsets (units of one 512-half fragment = 1KB)
#define FOFF_L1 0     // w1  A-layout: M=32, K=32(pad 6)   -> 2  frags
#define FOFF_L2 2     // w2  A-layout: M=128, K=32         -> 8  frags
#define FOFF_L3 10    // w3  B-layout: K=128, N=256        -> 64 frags
#define FOFF_F1 74    // fw1 B-layout: K=256, N=128        -> 64 frags
#define FOFF_F2 138   // fw2 B-layout: K=128, N=64         -> 16 frags
#define FOFF_F3 154   // fw3 B-layout: K=64,  N=80(pad 75) -> 10 frags
#define NFRAGS  164

union AFrag {
  v16h h;
  uint4 q[2];
  unsigned int u[8];
  _Float16 e[16];
};

union H8 {               // 8 packed f16 = one b128 store / half an A fragment
  uint4 q;
  _Float16 e[8];
};

__device__ __forceinline__ v8f wmma16(v16h a, v16h b, v8f c) {
  // D = A(16x32 f16) * B(32x16 f16) + C(16x16 f32)
  return __builtin_amdgcn_wmma_f32_16x16x32_f16(
      /*neg_a=*/false, a, /*neg_b=*/false, b,
      /*c_mod=*/(short)0, c, /*reuse_a=*/false, /*reuse_b=*/false);
}

// A-fragment gather from an LDS activation tile (row-major [16 x cin] f16).
// 16-bit A 16x32 layout: lane<16 : M=lane, K={ks*32+0..7, ks*32+16..23}
//                        lane>=16: M=lane-16, K shifted by +8.
__device__ __forceinline__ v16h load_a_lds(const _Float16* buf, int cin,
                                           int ks, int m, int hi) {
  const unsigned int* p = (const unsigned int*)buf;
  int b1 = (m * cin + ks * 32 + hi * 8) >> 1;  // u32 index, 16B aligned
  AFrag f;
  f.q[0] = *(const uint4*)(p + b1);
  f.q[1] = *(const uint4*)(p + b1 + 8);        // +16 halves (K+16)
  return f.h;
}

// Fragment load from the repacked global table (per-lane 32B contiguous).
__device__ __forceinline__ v16h load_frag(const _Float16* frags, int fragIdx,
                                          int lane) {
  const uint4* p = (const uint4*)(frags + (size_t)(fragIdx * 32 + lane) * 16);
  AFrag f;
  f.q[0] = p[0];
  f.q[1] = p[1];
  return f.h;
}

// ---------------------------------------------------------------------------
// Stage 0: zero the segment-max accumulator (init 0 == maximum(seg, 0)).
// ---------------------------------------------------------------------------
__global__ void pn_zero_kernel(float* __restrict__ p, int n) {
  int i = blockIdx.x * blockDim.x + threadIdx.x;
  if (i < n) p[i] = 0.0f;
}

// ---------------------------------------------------------------------------
// Stage 1: repack W[out][in] (f32 row-major) into f16 WMMA fragments.
// alayout=0 (contiguous K): K = ks*32 + h + hi*16,             col = nt*16+lane%16
// alayout=1 (A 16x32): K = ks*32 + (h<8?h:h+8) + hi*8,    row = nt*16+lane%16
// Zero-pad K>=Cin and row/col>=Cout.
// ---------------------------------------------------------------------------
__global__ void pn_repack_kernel(const float* __restrict__ W, int Cin, int Cout,
                                 _Float16* __restrict__ dst, int ntN,
                                 int alayout) {
  int lane = threadIdx.x;          // 0..31
  int frag = blockIdx.x;           // ks*ntN + nt
  int ks = frag / ntN;
  int nt = frag - ks * ntN;
  int row = nt * 16 + (lane & 15); // out-channel (A: M, B: N)
  int hi = lane >> 4;
#pragma unroll
  for (int h = 0; h < 16; h++) {
    int kl = alayout ? ((h < 8 ? h : h + 8) + hi * 8) : (h + hi * 16);
    int k = ks * 32 + kl;
    float v = (k < Cin && row < Cout) ? W[row * Cin + k] : 0.0f;
    dst[((size_t)frag * 32 + lane) * 16 + h] = (_Float16)v;
  }
}

// ---------------------------------------------------------------------------
// Stage 2: per-point MLP 6->32->128->256 + fused segment-max (sorted batch).
// One wave == 2 consecutive 16-point M-tiles (32 points).
// ---------------------------------------------------------------------------
#define WAVES1 4
#define MT     2
__global__ void __launch_bounds__(32 * WAVES1, 4)  // >=4 waves/EU -> <=256 VGPR
pn_point_kernel(const float* __restrict__ points, const float* __restrict__ color,
                const long long* __restrict__ batch,
                const float* __restrict__ b1, const float* __restrict__ b2,
                const float* __restrict__ b3,
                const _Float16* __restrict__ frags,
                float* __restrict__ seg, int npts) {
  __shared__ __align__(16) _Float16 h1s[WAVES1][MT][16 * 32];   // 8 KB total

  const int wid  = threadIdx.x >> 5;
  const int lane = threadIdx.x & 31;
  const int grp  = blockIdx.x * WAVES1 + wid;  // 32-point group
  const int base = grp * 16 * MT;
  if (base >= npts) return;  // wave-uniform

  const int m  = lane & 15;
  const int hi = lane >> 4;

  // batch ids per M-tile (wave-uniform per row; batch is sorted).
  int  brow[MT], bfirst[MT];
  bool uni[MT];
#pragma unroll
  for (int t = 0; t < MT; t++) {
    brow[t]   = (int)batch[base + t * 16 + m];
    bfirst[t] = __shfl(brow[t], 0, 32);
    uni[t]    = (bfirst[t] == __shfl(brow[t], 15, 32));
  }

  // weight A-fragments for layers 1 (2x) and 2 (8x), reused by both tiles
  v16h w1a[2], w2a[8];
#pragma unroll
  for (int mt = 0; mt < 2; mt++) w1a[mt] = load_frag(frags, FOFF_L1 + mt, lane);
#pragma unroll
  for (int mt = 0; mt < 8; mt++) w2a[mt] = load_frag(frags, FOFF_L2 + mt, lane);

  // ---- layers 1 and 2 per tile: weights in A, activations in B ----
  // D layout: lane = point (N), VGPR v = out-channel v + 8*hi (M)
  // -> each lane's 8 outputs are 8 contiguous channels of its point.
  AFrag a3[MT][4];  // layer-3 A fragments, filled directly by layer-2 outputs
#pragma unroll
  for (int t = 0; t < MT; t++) {
    // input B fragment: [6 x 16pts] zero-padded to [32 x 16]
    // lane<16: point=lane, K={0..7,16..23} (features in halves 0..5); lane>=16: pad
    AFrag bin;
#pragma unroll
    for (int i = 0; i < 8; i++) bin.u[i] = 0u;
    if (lane < 16) {
      int p = base + t * 16 + lane;
      bin.e[0] = (_Float16)points[p * 3 + 0];
      bin.e[1] = (_Float16)points[p * 3 + 1];
      bin.e[2] = (_Float16)points[p * 3 + 2];
      bin.e[3] = (_Float16)color[p * 3 + 0];
      bin.e[4] = (_Float16)color[p * 3 + 1];
      bin.e[5] = (_Float16)color[p * 3 + 2];
    }

    // layer 1: 6->32, relu -> LDS (hand-off genuinely crosses lanes)
#pragma unroll
    for (int mt = 0; mt < 2; mt++) {
      v8f c = {};
      c = wmma16(w1a[mt], bin.h, c);
      const float* bv = b1 + mt * 16 + 8 * hi;  // 8 consecutive biases
      H8 o;
#pragma unroll
      for (int v = 0; v < 8; v++)
        o.e[v] = (_Float16)fmaxf(c[v] + bv[v], 0.0f);
      *(uint4*)&h1s[wid][t][m * 32 + mt * 16 + 8 * hi] = o.q;
    }

    // layer 2: 32->128, relu.  B-act fragment = 32B contiguous per lane.
    AFrag ba;
    {
      const uint4* pb = (const uint4*)&h1s[wid][t][m * 32 + hi * 16];
      ba.q[0] = pb[0];
      ba.q[1] = pb[1];
    }
    // Layer-2 packed outputs ARE layer-3 A-fragment halves:
    //   a3[t][ks].halves 0..7  = channels 32ks +  8hi + 0..7     (mt = 2ks)
    //   a3[t][ks].halves 8..15 = channels 32ks + 16 + 8hi + 0..7 (mt = 2ks+1)
#pragma unroll
    for (int mt = 0; mt < 8; mt++) {
      v8f c = {};
      c = wmma16(w2a[mt], ba.h, c);
      const float* bv = b2 + mt * 16 + 8 * hi;
      H8 o;
#pragma unroll
      for (int v = 0; v < 8; v++)
        o.e[v] = (_Float16)fmaxf(c[v] + bv[v], 0.0f);
      a3[t][mt >> 1].q[mt & 1] = o.q;
    }
  }

  // ---- layer 3: 128->256 (no relu), act in A / weights in B,
  //      D layout lane=channel -> cheap lane-parallel segment max ----
#pragma unroll
  for (int nt = 0; nt < 16; nt++) {
    v8f acc[MT];
#pragma unroll
    for (int t = 0; t < MT; t++) acc[t] = (v8f){};
#pragma unroll
    for (int ks = 0; ks < 4; ks++) {
      v16h bf = load_frag(frags, FOFF_L3 + ks * 16 + nt, lane);
#pragma unroll
      for (int t = 0; t < MT; t++) acc[t] = wmma16(a3[t][ks].h, bf, acc[t]);
    }
    float bias = b3[nt * 16 + m];

#pragma unroll
    for (int t = 0; t < MT; t++) {
      if (uni[t]) {
        // tile is one segment: in-wave 16-row max, one atomic per channel
        float mx = fmaxf(acc[t][0] + bias, 0.0f);
#pragma unroll
        for (int v = 1; v < 8; v++) mx = fmaxf(mx, acc[t][v] + bias);
        mx = fmaxf(mx, __shfl_xor(mx, 16, 32));
        if (lane < 16)
          atomicMax((int*)&seg[(size_t)bfirst[t] * 256 + nt * 16 + lane],
                    __float_as_int(mx));  // vals >= 0: int order == float order
      } else {
        // segment boundary inside tile: per-row atomics
#pragma unroll
        for (int v = 0; v < 8; v++) {
          float x = fmaxf(acc[t][v] + bias, 0.0f);
          int bb = __shfl(brow[t], v + hi * 8, 32);
          atomicMax((int*)&seg[(size_t)bb * 256 + nt * 16 + m],
                    __float_as_int(x));
        }
      }
    }
  }
}

// ---------------------------------------------------------------------------
// Stage 3: FC head 256->128->64->75 with fused coef_std/coef_mean affine.
// One wave == one 16-segment tile (4096/16 = 256 tiles).
// ---------------------------------------------------------------------------
#define WAVES2 8
__global__ void __launch_bounds__(32 * WAVES2)
pn_head_kernel(const float* __restrict__ seg,
               const float* __restrict__ fb1, const float* __restrict__ fb2,
               const float* __restrict__ fb3,
               const float* __restrict__ coef_mean,
               const float* __restrict__ coef_std,
               const _Float16* __restrict__ frags, float* __restrict__ out) {
  __shared__ __align__(16) _Float16 g1s[WAVES2][16 * 128];
  __shared__ __align__(16) _Float16 g2s[WAVES2][16 * 64];

  const int wid  = threadIdx.x >> 5;
  const int lane = threadIdx.x & 31;
  const int tile = blockIdx.x * WAVES2 + wid;  // 0..255
  const int row0 = tile * 16;
  const int m  = lane & 15;
  const int hi = lane >> 4;

  // ---- A fragments for fc1 directly from seg (f32 -> f16), 8 k-steps ----
  v16h aIn[8];
#pragma unroll
  for (int ks = 0; ks < 8; ks++) {
    const float* p = seg + (size_t)(row0 + m) * 256 + ks * 32 + hi * 8;
    AFrag f;
#pragma unroll
    for (int i = 0; i < 8; i++) {
      f.e[i]     = (_Float16)p[i];        // K = ks*32 + hi*8 + i
      f.e[8 + i] = (_Float16)p[16 + i];   // K = ks*32 + 16 + hi*8 + i
    }
    aIn[ks] = f.h;
  }

  // ---- fc1: 256->128, relu ----
#pragma unroll
  for (int nt = 0; nt < 8; nt++) {
    v8f c = {};
#pragma unroll
    for (int ks = 0; ks < 8; ks++)
      c = wmma16(aIn[ks], load_frag(frags, FOFF_F1 + ks * 8 + nt, lane), c);
    float bias = fb1[nt * 16 + m];
#pragma unroll
    for (int v = 0; v < 8; v++) {
      float x = fmaxf(c[v] + bias, 0.0f);
      g1s[wid][(v + hi * 8) * 128 + nt * 16 + m] = (_Float16)x;
    }
  }

  // ---- fc2: 128->64, relu ----
  v16h a2[4];
#pragma unroll
  for (int ks = 0; ks < 4; ks++) a2[ks] = load_a_lds(g1s[wid], 128, ks, m, hi);
#pragma unroll
  for (int nt = 0; nt < 4; nt++) {
    v8f c = {};
#pragma unroll
    for (int ks = 0; ks < 4; ks++)
      c = wmma16(a2[ks], load_frag(frags, FOFF_F2 + ks * 4 + nt, lane), c);
    float bias = fb2[nt * 16 + m];
#pragma unroll
    for (int v = 0; v < 8; v++) {
      float x = fmaxf(c[v] + bias, 0.0f);
      g2s[wid][(v + hi * 8) * 64 + nt * 16 + m] = (_Float16)x;
    }
  }

  // ---- fc3: 64->80 (75 valid), affine, store ----
  v16h a3h[2];
#pragma unroll
  for (int ks = 0; ks < 2; ks++) a3h[ks] = load_a_lds(g2s[wid], 64, ks, m, hi);
#pragma unroll
  for (int nt = 0; nt < 5; nt++) {
    v8f c = {};
#pragma unroll
    for (int ks = 0; ks < 2; ks++)
      c = wmma16(a3h[ks], load_frag(frags, FOFF_F3 + ks * 5 + nt, lane), c);
    int n = nt * 16 + m;
    if (n < KOUT) {
      float bias = fb3[n];
      float sc = coef_std[n];
      float mu = coef_mean[n];
#pragma unroll
      for (int v = 0; v < 8; v++) {
        float h = c[v] + bias;  // no relu on last layer
        out[(size_t)(row0 + v + hi * 8) * KOUT + n] = h * sc + mu;
      }
    }
  }
}

// ---------------------------------------------------------------------------
// Launch. Workspace layout: [ seg f32 4096*256 | f16 fragment table 164KB ].
// ---------------------------------------------------------------------------
extern "C" void kernel_launch(void* const* d_in, const int* in_sizes, int n_in,
                              void* d_out, int out_size, void* d_ws, size_t ws_size,
                              hipStream_t stream) {
  const float*     points = (const float*)d_in[0];
  const float*     color  = (const float*)d_in[1];
  const long long* batch  = (const long long*)d_in[2];
  const float* w1  = (const float*)d_in[3];  const float* b1  = (const float*)d_in[4];
  const float* w2  = (const float*)d_in[5];  const float* b2  = (const float*)d_in[6];
  const float* w3  = (const float*)d_in[7];  const float* b3  = (const float*)d_in[8];
  const float* fw1 = (const float*)d_in[9];  const float* fb1 = (const float*)d_in[10];
  const float* fw2 = (const float*)d_in[11]; const float* fb2 = (const float*)d_in[12];
  const float* fw3 = (const float*)d_in[13]; const float* fb3 = (const float*)d_in[14];
  const float* coef_mean = (const float*)d_in[15];
  const float* coef_std  = (const float*)d_in[16];
  float* out = (float*)d_out;

  float*    seg   = (float*)d_ws;                                   // 4 MB
  _Float16* frags = (_Float16*)((char*)d_ws + (size_t)NSEG * 256 * 4);

  (void)in_sizes; (void)n_in; (void)out_size; (void)ws_size;

  // 0) zero segment-max accumulator (re-done every launch; graph-safe)
  pn_zero_kernel<<<(NSEG * 256 + 255) / 256, 256, 0, stream>>>(seg, NSEG * 256);

  // 1) repack weights: w2 in contiguous K order (alayout 0), the rest in table K order (alayout 1)
  pn_repack_kernel<<< 2, 32, 0, stream>>>(w1,    6,  32, frags + (size_t)FOFF_L1 * 512,  2, 1);
  pn_repack_kernel<<< 8, 32, 0, stream>>>(w2,   32, 128, frags + (size_t)FOFF_L2 * 512,  8, 0);
  pn_repack_kernel<<<64, 32, 0, stream>>>(w3,  128, 256, frags + (size_t)FOFF_L3 * 512, 16, 1);
  pn_repack_kernel<<<64, 32, 0, stream>>>(fw1, 256, 128, frags + (size_t)FOFF_F1 * 512,  8, 1);
  pn_repack_kernel<<<16, 32, 0, stream>>>(fw2, 128,  64, frags + (size_t)FOFF_F2 * 512,  4, 1);
  pn_repack_kernel<<<10, 32, 0, stream>>>(fw3,  64, KOUT, frags + (size_t)FOFF_F3 * 512, 5, 1);

  // 2) per-point MLP + segment max: 31250 groups of 32 points, 4 waves/block
  int groups  = (NPTS + 16 * MT - 1) / (16 * MT);
  int blocks1 = (groups + WAVES1 - 1) / WAVES1;
  pn_point_kernel<<<blocks1, 32 * WAVES1, 0, stream>>>(
      points, color, batch, b1, b2, b3, frags, seg, NPTS);

  // 3) FC head: 256 tiles, 8 waves/block -> 32 blocks
  pn_head_kernel<<<(NSEG / 16 + WAVES2 - 1) / WAVES2, 32 * WAVES2, 0, stream>>>(
      seg, fb1, fb2, fb3, coef_mean, coef_std, frags, out);
}
